// Channel_Self_Attention_42339787604328
// MI455X (gfx1250) — compile-verified, matrix operand order checked by reading, not run on hardware
//
#include <hip/hip_runtime.h>

#define NBATCH 16
#define NCHAN  128
#define NHW    36864
#define KC     64
#define S1     72    // LDS row stride (elems) phase 1: 144B rows -> 16B aligned frags
#define S3     136   // LDS row stride (elems) phase 3: 272B rows -> 16B aligned frags

typedef __bf16 bf16x16 __attribute__((ext_vector_type(16)));
typedef float  v8f     __attribute__((ext_vector_type(8)));

struct FragBits { uint4 lo; uint4 hi; };

__device__ __forceinline__ unsigned short f2bf(float f) {
  unsigned int u = __float_as_uint(f);
  u += 0x7FFFu + ((u >> 16) & 1u);       // round-to-nearest-even
  return (unsigned short)(u >> 16);
}

// Loads one 16x32 bf16 WMMA operand fragment from a row-major bf16 buffer.
// Works for BOTH the A operand (rows = M) and the B operand when the source is
// the transpose-view (rows = N), since A and B K-striping patterns match:
// lane L: row = base + (L&15), elems[0..7] = K[8*hi .. 8*hi+7],
//         elems[8..15] = K[8*hi+16 .. 8*hi+23], hi = L>>4.
__device__ __forceinline__ bf16x16 ld_frag(const unsigned short* base, int rowStride,
                                           int rowBase, int kbase, int lane) {
  const int r  = rowBase + (lane & 15);
  const int hi = lane >> 4;
  const unsigned short* p = base + r * rowStride + kbase + 8 * hi;
  FragBits fb;
  fb.lo = *reinterpret_cast<const uint4*>(p);
  fb.hi = *reinterpret_cast<const uint4*>(p + 16);
  return __builtin_bit_cast(bf16x16, fb);
}

__global__ void zero_kernel(float* __restrict__ p) {
  p[(size_t)blockIdx.x * 256 + threadIdx.x] = 0.0f;
}

// ---------------- Phase 1: energy[b] = Q Q^T (partial K, atomic reduce) ------
__global__ void __launch_bounds__(256)
gram_kernel(const float* __restrict__ x, float* __restrict__ energy, int chunksPerWG) {
  __shared__ unsigned short qlds[NCHAN * S1];
  const int t    = threadIdx.x;
  const int lane = t & 31;
  const int w    = t >> 5;               // wave id 0..7 -> output tile-row
  const int b    = blockIdx.y;
  const float* xb = x + (size_t)b * NCHAN * NHW;

  v8f acc[8];
#pragma unroll
  for (int j = 0; j < 8; ++j) acc[j] = (v8f){0.f,0.f,0.f,0.f,0.f,0.f,0.f,0.f};

  const int row  = t >> 1;               // 0..127
  const int col0 = (t & 1) * 32;

  for (int ch = 0; ch < chunksPerWG; ++ch) {
    const int nbase = (blockIdx.x * chunksPerWG + ch) * KC;
    // stage 128 x 64 f32 -> bf16 LDS
    const float4* src = reinterpret_cast<const float4*>(xb + (size_t)row * NHW + nbase + col0);
    unsigned short* dst = qlds + row * S1 + col0;
#pragma unroll
    for (int i = 0; i < 8; ++i) {
      float4 v = src[i];
      uint2 pk;
      pk.x = (unsigned)f2bf(v.x) | ((unsigned)f2bf(v.y) << 16);
      pk.y = (unsigned)f2bf(v.z) | ((unsigned)f2bf(v.w) << 16);
      *reinterpret_cast<uint2*>(dst + i * 4) = pk;
    }
    __syncthreads();
#pragma unroll
    for (int ks = 0; ks < 2; ++ks) {
      const int kb = ks * 32;
      // Preload A + all 8 B fragments so ~18 ds_load_b128 are in flight
      // before the first WMMA consumes them (partial dscnt waits).
      bf16x16 A = ld_frag(qlds, S1, 16 * w, kb, lane);
      bf16x16 Bfr[8];
#pragma unroll
      for (int j = 0; j < 8; ++j)
        Bfr[j] = ld_frag(qlds, S1, 16 * j, kb, lane);  // B = Q^T tile == same layout
#pragma unroll
      for (int j = 0; j < 8; ++j)
        acc[j] = __builtin_amdgcn_wmma_f32_16x16x32_bf16(
            false, A, false, Bfr[j], (short)0, acc[j], false, false);
    }
    __syncthreads();
  }

  float* eb = energy + (size_t)b * NCHAN * NCHAN;
  const int col = lane & 15;
  const int h8  = (lane >> 4) * 8;
#pragma unroll
  for (int j = 0; j < 8; ++j)
#pragma unroll
    for (int r = 0; r < 8; ++r) {
      const int cr = 16 * w + r + h8;           // D layout: M = r + 8*hi
      atomicAdd(eb + cr * NCHAN + 16 * j + col, acc[j][r]);
    }
}

// ---------------- Phase 2: row softmax, emit bf16 attn ----------------------
__global__ void __launch_bounds__(256)
softmax_kernel(const float* __restrict__ energy, unsigned short* __restrict__ attn) {
  const int t    = threadIdx.x;
  const int lane = t & 31;
  const int row  = blockIdx.x * 8 + (t >> 5);   // 0..2047
  const float* e = energy + (size_t)row * NCHAN;
  float v[4];
#pragma unroll
  for (int i = 0; i < 4; ++i) v[i] = e[lane + 32 * i];
  float m = fmaxf(fmaxf(v[0], v[1]), fmaxf(v[2], v[3]));
#pragma unroll
  for (int off = 16; off > 0; off >>= 1) m = fmaxf(m, __shfl_xor(m, off, 32));
  float s = 0.f;
#pragma unroll
  for (int i = 0; i < 4; ++i) { v[i] = __expf(v[i] - m); s += v[i]; }
#pragma unroll
  for (int off = 16; off > 0; off >>= 1) s += __shfl_xor(s, off, 32);
  const float inv = 1.0f / s;
#pragma unroll
  for (int i = 0; i < 4; ++i)
    attn[(size_t)row * NCHAN + lane + 32 * i] = f2bf(v[i] * inv);
}

// ---------------- Phase 3: out = gamma * (attn @ Q) + x ---------------------
__global__ void __launch_bounds__(256)
attn_out_kernel(const float* __restrict__ x, const unsigned short* __restrict__ attn,
                const float* __restrict__ gamma, float* __restrict__ out) {
  __shared__ unsigned short qT[NCHAN * S3];     // [n_local 0..127][d 0..127 (+pad)]
  const int t     = threadIdx.x;
  const int lane  = t & 31;
  const int w     = t >> 5;
  const int b     = blockIdx.y;
  const int nbase = blockIdx.x * 128;
  const float* xb = x + (size_t)b * NCHAN * NHW;

  // stage q tile TRANSPOSED: thread owns d0..d0+3 for n = 16*ng..16*ng+15
  const int ng = t & 7;
  const int d0 = (t >> 3) * 4;
#pragma unroll
  for (int i = 0; i < 4; ++i) {
    const int nb = nbase + 16 * ng + 4 * i;
    float4 r0 = *reinterpret_cast<const float4*>(xb + (size_t)(d0 + 0) * NHW + nb);
    float4 r1 = *reinterpret_cast<const float4*>(xb + (size_t)(d0 + 1) * NHW + nb);
    float4 r2 = *reinterpret_cast<const float4*>(xb + (size_t)(d0 + 2) * NHW + nb);
    float4 r3 = *reinterpret_cast<const float4*>(xb + (size_t)(d0 + 3) * NHW + nb);
    const float* p0 = reinterpret_cast<const float*>(&r0);
    const float* p1 = reinterpret_cast<const float*>(&r1);
    const float* p2 = reinterpret_cast<const float*>(&r2);
    const float* p3 = reinterpret_cast<const float*>(&r3);
#pragma unroll
    for (int nn = 0; nn < 4; ++nn) {
      uint2 pk;
      pk.x = (unsigned)f2bf(p0[nn]) | ((unsigned)f2bf(p1[nn]) << 16);
      pk.y = (unsigned)f2bf(p2[nn]) | ((unsigned)f2bf(p3[nn]) << 16);
      *reinterpret_cast<uint2*>(qT + (16 * ng + 4 * i + nn) * S3 + d0) = pk;
    }
  }
  __syncthreads();

  v8f acc[8];
#pragma unroll
  for (int j = 0; j < 8; ++j) acc[j] = (v8f){0.f,0.f,0.f,0.f,0.f,0.f,0.f,0.f};

  const unsigned short* ab = attn + (size_t)b * NCHAN * NCHAN;
#pragma unroll
  for (int ks = 0; ks < 4; ++ks) {
    const int kb = 32 * ks;
    bf16x16 A = ld_frag(ab, NCHAN, 16 * w, kb, lane);   // attn from global (L2-hot)
    bf16x16 Bfr[8];
#pragma unroll
    for (int j = 0; j < 8; ++j)
      Bfr[j] = ld_frag(qT, S3, 16 * j, kb, lane);       // transposed view == B layout
#pragma unroll
    for (int j = 0; j < 8; ++j)
      acc[j] = __builtin_amdgcn_wmma_f32_16x16x32_bf16(
          false, A, false, Bfr[j], (short)0, acc[j], false, false);
  }

  const float g   = gamma[0];
  const int   col = lane & 15;
  const int   h8  = (lane >> 4) * 8;
#pragma unroll
  for (int j = 0; j < 8; ++j)
#pragma unroll
    for (int r = 0; r < 8; ++r) {
      const int c = 16 * w + r + h8;
      const size_t idx = (size_t)b * NCHAN * NHW + (size_t)c * NHW + (nbase + 16 * j + col);
      out[idx] = g * acc[j][r] + x[idx];    // residual re-read hits L2 (same tile)
    }
}

extern "C" void kernel_launch(void* const* d_in, const int* in_sizes, int n_in,
                              void* d_out, int out_size, void* d_ws, size_t ws_size,
                              hipStream_t stream) {
  (void)in_sizes; (void)n_in; (void)out_size; (void)ws_size;
  const float* x     = (const float*)d_in[0];
  const float* gamma = (const float*)d_in[1];
  float* out = (float*)d_out;

  float* energy = (float*)d_ws;                                              // 1 MB f32
  unsigned short* attn =
      (unsigned short*)((char*)d_ws + (size_t)NBATCH * NCHAN * NCHAN * 4);   // 0.5 MB bf16

  zero_kernel<<<dim3(NBATCH * NCHAN * NCHAN / 256), 256, 0, stream>>>(energy);
  gram_kernel<<<dim3(48, NBATCH), 256, 0, stream>>>(x, energy, /*chunksPerWG=*/12);
  softmax_kernel<<<dim3(NBATCH * NCHAN / 8), 256, 0, stream>>>(energy, attn);
  attn_out_kernel<<<dim3(NHW / 128, NBATCH), 256, 0, stream>>>(x, attn, gamma, out);
}
